// MultiHeadAttention_23167053594932
// MI455X (gfx1250) — matrix operand order repaired and checked by reading, not compiled, not run on hardware
//
#include <hip/hip_runtime.h>

// Problem constants (from reference): B=2, S=2048, D=1024, H=16, dk=64
#define BATCH   2
#define S_LEN   2048
#define DMODEL  1024
#define NHEADS  16
#define DK      64
#define D3      3072   // 3*DMODEL

typedef __bf16 bf16;
typedef __attribute__((ext_vector_type(16))) __bf16 v16bf;
typedef __attribute__((ext_vector_type(8)))  __bf16 bf16x8;
typedef __attribute__((ext_vector_type(8)))  float  v8f;

union V16 { v16bf v; bf16x8 h[2]; };

static __device__ __forceinline__ bf16x8 load8(const bf16* p) {
  return *(const bf16x8*)p;   // 16-byte load; all call sites are 16B aligned
}

static __device__ __forceinline__ v8f wmma_bf16(const V16& a, const V16& b, v8f c) {
  // D = A(16x32 bf16) * B(32x16 bf16) + C(16x16 f32)
  return __builtin_amdgcn_wmma_f32_16x16x32_bf16(
      /*neg_a=*/false, a.v, /*neg_b=*/false, b.v,
      /*c_mod=*/(short)0, c, /*reuse_a=*/false, /*reuse_b=*/false);
}

// ---------------------------------------------------------------------------
// 1) fp32 -> bf16 conversion (4 elements / thread; sizes are multiples of 1024)
// ---------------------------------------------------------------------------
__global__ void __launch_bounds__(256) cvt_f32_to_bf16_x4(
    const float* __restrict__ in, bf16* __restrict__ out) {
  const int i = (blockIdx.x * 256 + threadIdx.x) * 4;
  const float4 v = *(const float4*)(in + i);
  out[i + 0] = (bf16)v.x;
  out[i + 1] = (bf16)v.y;
  out[i + 2] = (bf16)v.z;
  out[i + 3] = (bf16)v.w;
}

// ---------------------------------------------------------------------------
// 2) GEMM: C[M,N] = A[M,K] * W[N,K]^T  (bf16 operands, fp32 accumulate)
//    256 threads = 8 waves as 4(m) x 2(n); wave tile 32x64 = 8 accumulators;
//    block tile 128x128.  K-loop is register double-buffered so WMMAs overlap
//    the next tile's global loads (waits become s_wait_loadcnt N, not 0).
// ---------------------------------------------------------------------------
struct GTile { V16 a[2]; V16 b[4]; };

template <bool OUT_F32>
__global__ void __launch_bounds__(256) gemm_bf16_rowB(
    const bf16* __restrict__ A, const bf16* __restrict__ W,
    void* __restrict__ Cout, int M, int N, int K) {
  const int lane = threadIdx.x & 31;
  const int wid  = threadIdx.x >> 5;
  const int hi   = lane >> 4;       // half-wave: 0 or 1
  const int l16  = lane & 15;

  const int m_base = blockIdx.x * 128 + (wid & 3) * 32;
  const int n_base = blockIdx.y * 128 + (wid >> 2) * 64;

  const bf16* ar[2];
  ar[0] = A + (size_t)(m_base + l16) * K;
  ar[1] = ar[0] + (size_t)16 * K;
  const bf16* wr[4];
  wr[0] = W + (size_t)(n_base + l16) * K;
  wr[1] = wr[0] + (size_t)16 * K;
  wr[2] = wr[0] + (size_t)32 * K;
  wr[3] = wr[0] + (size_t)48 * K;

  auto load_tile = [&](GTile& t, int k) {
    // A-operand: lanes<16 hold K {k..k+7, k+16..k+23}; lanes>=16 hold {+8,+24}
    const int aoff = k + 8 * hi;
#pragma unroll
    for (int i = 0; i < 2; ++i) {
      t.a[i].h[0] = load8(ar[i] + aoff);
      t.a[i].h[1] = load8(ar[i] + aoff + 16);
    }
    // B-operand: lanes<16 hold K {k..k+7, k+16..k+23}; lanes>=16 hold {+8,+24}
    const int boff = k + 8 * hi;
#pragma unroll
    for (int j = 0; j < 4; ++j) {
      t.b[j].h[0] = load8(wr[j] + boff);
      t.b[j].h[1] = load8(wr[j] + boff + 16);
    }
  };

  v8f acc[2][4] = {};
  auto mma_tile = [&](const GTile& t) {
#pragma unroll
    for (int i = 0; i < 2; ++i)
#pragma unroll
      for (int j = 0; j < 4; ++j)
        acc[i][j] = wmma_bf16(t.a[i], t.b[j], acc[i][j]);
  };

  GTile t0, t1;
  load_tile(t0, 0);
#pragma unroll 1
  for (int k = 0; k + 64 < K; k += 64) {
    load_tile(t1, k + 32);   // prefetch while t0's MMAs run
    mma_tile(t0);
    load_tile(t0, k + 64);   // prefetch while t1's MMAs run
    mma_tile(t1);
  }
  load_tile(t1, K - 32);
  mma_tile(t0);
  mma_tile(t1);

  // C/D layout: N = lane%16, M = vgpr + 8*(lane>=16)
#pragma unroll
  for (int i = 0; i < 2; ++i) {
#pragma unroll
    for (int j = 0; j < 8; ++j) {
      const int r  = m_base + 16 * i + j + 8 * hi;
      const int n0 = n_base + l16;
      if constexpr (OUT_F32) {
        float* C = (float*)Cout;
        C[(size_t)r * N + n0]      = acc[i][0][j];
        C[(size_t)r * N + n0 + 16] = acc[i][1][j];
        C[(size_t)r * N + n0 + 32] = acc[i][2][j];
        C[(size_t)r * N + n0 + 48] = acc[i][3][j];
      } else {
        bf16* C = (bf16*)Cout;
        C[(size_t)r * N + n0]      = (bf16)acc[i][0][j];
        C[(size_t)r * N + n0 + 16] = (bf16)acc[i][1][j];
        C[(size_t)r * N + n0 + 32] = (bf16)acc[i][2][j];
        C[(size_t)r * N + n0 + 48] = (bf16)acc[i][3][j];
      }
    }
  }
}

// ---------------------------------------------------------------------------
// 3) RoPE + head split. One thread per (b, s, h, pair) ; pair = dk/2 = 32.
//    Writes Qh, Kh as [B,H,S,DK] bf16 and V transposed as [B,H,DK,S] bf16.
// ---------------------------------------------------------------------------
__global__ void __launch_bounds__(256) rope_split_kernel(
    const bf16* __restrict__ qkv, bf16* __restrict__ Qh,
    bf16* __restrict__ Kh, bf16* __restrict__ Vt) {
  const int tid = blockIdx.x * 256 + threadIdx.x;  // exactly B*S*H*32 threads
  const int p = tid & 31;
  const int h = (tid >> 5) & (NHEADS - 1);
  const int s = (tid >> 9) & (S_LEN - 1);
  const int b = tid >> 20;                          // 512*2048 = 2^20 per batch

  const size_t row = ((size_t)b * S_LEN + s) * D3;
  const int col = h * DK + 2 * p;
  const float q1 = (float)qkv[row + col];
  const float q2 = (float)qkv[row + col + 1];
  const float k1 = (float)qkv[row + DMODEL + col];
  const float k2 = (float)qkv[row + DMODEL + col + 1];
  const float v1 = (float)qkv[row + 2 * DMODEL + col];
  const float v2 = (float)qkv[row + 2 * DMODEL + col + 1];

  // inv_freq = 10000^(-2p/64) = exp(-2p/64 * ln(10000))
  const float inv_freq = __expf(-(float)(2 * p) * (9.210340371976184f / 64.0f));
  const float ang = (float)s * inv_freq;
  float sn, cs;
  __sincosf(ang, &sn, &cs);

  const size_t qk_idx = (((size_t)(b * NHEADS + h) * S_LEN) + s) * DK + 2 * p;
  Qh[qk_idx]     = (bf16)(q1 * cs - q2 * sn);
  Qh[qk_idx + 1] = (bf16)(q1 * sn + q2 * cs);
  Kh[qk_idx]     = (bf16)(k1 * cs - k2 * sn);
  Kh[qk_idx + 1] = (bf16)(k1 * sn + k2 * cs);

  const size_t v_idx = ((size_t)(b * NHEADS + h) * DK + 2 * p) * S_LEN + s;
  Vt[v_idx]         = (bf16)v1;
  Vt[v_idx + S_LEN] = (bf16)v2;
}

// ---------------------------------------------------------------------------
// 4) Causal flash attention.  One wave owns a 32-query tile (two 16-query
//    groups sharing each K/V fetch), keys stepped by 32.
//    S' = K_tile * Q^T  -> query lives in lane index (softmax = shfl_xor 16),
//    keys live across accumulator VGPRs, which maps 1:1 onto the A-operand
//    layout for P*V  ->  no LDS, no cross-lane transpose.
//    Pipelined: V loads issued first, next K tile prefetched before the
//    softmax VALU chain, so loads overlap VALU + WMMA work.
// ---------------------------------------------------------------------------
__global__ void __launch_bounds__(256) flash_attn_kernel(
    const bf16* __restrict__ Qh, const bf16* __restrict__ Kh,
    const bf16* __restrict__ Vt, bf16* __restrict__ attn_out) {
  const int lane = threadIdx.x & 31;
  const int wid  = threadIdx.x >> 5;
  const int hi   = lane >> 4;
  const int l16  = lane & 15;

  const int bh = blockIdx.x >> 3;            // (b*H + h), 8 blocks per bh
  const int b  = bh >> 4;                    // H == 16
  const int h  = bh & (NHEADS - 1);
  // strided q-tile assignment balances causal work across blocks
  const int qt = (blockIdx.x & 7) + 8 * wid; // 0..63
  const int q0 = qt * 32;

  const bf16* Qbase = Qh + (size_t)bh * S_LEN * DK;
  V16 bq[2][2];
#pragma unroll
  for (int g = 0; g < 2; ++g) {
    const bf16* qrow = Qbase + (size_t)(q0 + 16 * g + l16) * DK;
    const int b0 = 8 * hi;
    bq[g][0].h[0] = load8(qrow + b0);       bq[g][0].h[1] = load8(qrow + b0 + 16);
    bq[g][1].h[0] = load8(qrow + 32 + b0);  bq[g][1].h[1] = load8(qrow + 32 + b0 + 16);
  }

  float m_run[2] = {-1e30f, -1e30f};
  float l_run[2] = {0.0f, 0.0f};
  v8f o[2][4] = {};

  const bf16* Kbase = Kh + (size_t)bh * S_LEN * DK;
  const bf16* Vbase = Vt + (size_t)bh * DK * S_LEN;
  const int ntiles = ((q0 + 31) >> 5) + 1;   // causal: keys 0 .. q0+31

  // current K tile fragments (two 16-key sub-tiles x two 32-deep d chunks)
  V16 ka0, kb0, ka1, kb1;
  auto loadK = [&](int k0) {
    const bf16* krow0 = Kbase + (size_t)(k0 + l16) * DK;
    const bf16* krow1 = krow0 + 16 * DK;
    const int a0 = 8 * hi;
    ka0.h[0] = load8(krow0 + a0);       ka0.h[1] = load8(krow0 + a0 + 16);
    kb0.h[0] = load8(krow0 + 32 + a0);  kb0.h[1] = load8(krow0 + 32 + a0 + 16);
    ka1.h[0] = load8(krow1 + a0);       ka1.h[1] = load8(krow1 + a0 + 16);
    kb1.h[0] = load8(krow1 + 32 + a0);  kb1.h[1] = load8(krow1 + 32 + a0 + 16);
  };
  loadK(0);

#pragma unroll 1
  for (int t = 0; t < ntiles; ++t) {
    const int k0 = t * 32;

    // issue V loads early; consumed only after the softmax chain
    V16 bv[4];
#pragma unroll
    for (int c = 0; c < 4; ++c) {
      const bf16* vrow = Vbase + (size_t)(c * 16 + l16) * S_LEN + k0 + 8 * hi;
      bv[c].h[0] = load8(vrow);
      bv[c].h[1] = load8(vrow + 16);
    }

    // S' = K_tile(32x64) * Q^T per query group
    v8f s[2][2];
#pragma unroll
    for (int g = 0; g < 2; ++g) {
      v8f s0 = {}, s1 = {};
      s0 = wmma_bf16(ka0, bq[g][0], s0);
      s0 = wmma_bf16(kb0, bq[g][1], s0);
      s1 = wmma_bf16(ka1, bq[g][0], s1);
      s1 = wmma_bf16(kb1, bq[g][1], s1);
      s[g][0] = s0;
      s[g][1] = s1;
    }

    // prefetch next K tile; its latency hides under the softmax VALU chain
    if (t + 1 < ntiles) loadK(k0 + 32);

    // per-group online softmax; P packed directly into A-operand layout:
    // frag0 -> A vgprs 0-3 (keys 0-7 low / 8-15 high lanes),
    // frag1 -> A vgprs 4-7 (keys 16-23 / 24-31).  Zero cross-lane movement.
    V16 ap[2];
#pragma unroll
    for (int g = 0; g < 2; ++g) {
      const int q_glob = q0 + 16 * g + l16;
      float p0[8], p1[8];
      float mloc = -1e30f;
#pragma unroll
      for (int j = 0; j < 8; ++j) {
        const int key = k0 + j + 8 * hi;          // C layout: M = j + 8*hi
        float v0 = s[g][0][j] * 0.125f; if (key > q_glob)      v0 = -1e30f;
        float v1 = s[g][1][j] * 0.125f; if (key + 16 > q_glob) v1 = -1e30f;
        p0[j] = v0; p1[j] = v1;
        mloc = fmaxf(mloc, fmaxf(v0, v1));
      }
      mloc = fmaxf(mloc, __shfl_xor(mloc, 16, 32));
      const float m_new = fmaxf(m_run[g], mloc);
      const float alpha = __expf(m_run[g] - m_new);
      float rsum = 0.0f;
#pragma unroll
      for (int j = 0; j < 8; ++j) {
        const float e0 = __expf(p0[j] - m_new);
        const float e1 = __expf(p1[j] - m_new);
        rsum += e0 + e1;
        ap[g].v[j]     = (bf16)e0;
        ap[g].v[8 + j] = (bf16)e1;
      }
      rsum += __shfl_xor(rsum, 16, 32);
      l_run[g] = l_run[g] * alpha + rsum;
      m_run[g] = m_new;
      // rescale O (row q = j + 8*hi lives in lane j+8*hi's alpha)
#pragma unroll
      for (int j = 0; j < 8; ++j) {
        const float f = __shfl(alpha, j + 8 * hi, 32);
        o[g][0][j] *= f; o[g][1][j] *= f; o[g][2][j] *= f; o[g][3][j] *= f;
      }
    }

    // O += P(16x32) * V(32x16) per dk chunk (V pre-transposed -> contiguous)
#pragma unroll
    for (int g = 0; g < 2; ++g)
#pragma unroll
      for (int c = 0; c < 4; ++c)
        o[g][c] = wmma_bf16(ap[g], bv[c], o[g][c]);
  }

  // normalize and store merged heads: attn_out[b][s][h*64 + d] (bf16)
#pragma unroll
  for (int g = 0; g < 2; ++g) {
#pragma unroll
    for (int j = 0; j < 8; ++j) {
      const float linv = 1.0f / __shfl(l_run[g], j + 8 * hi, 32);
      const int srow = q0 + 16 * g + j + 8 * hi;
      bf16* dst = attn_out + ((size_t)b * S_LEN + srow) * DMODEL + h * DK + l16;
      dst[0]  = (bf16)(o[g][0][j] * linv);
      dst[16] = (bf16)(o[g][1][j] * linv);
      dst[32] = (bf16)(o[g][2][j] * linv);
      dst[48] = (bf16)(o[g][3][j] * linv);
    }
  }
}

// ---------------------------------------------------------------------------
// Launcher
// ---------------------------------------------------------------------------
extern "C" void kernel_launch(void* const* d_in, const int* in_sizes, int n_in,
                              void* d_out, int out_size, void* d_ws, size_t ws_size,
                              hipStream_t stream) {
  (void)in_sizes; (void)n_in; (void)out_size; (void)ws_size;

  const float* x     = (const float*)d_in[0];  // [B,S,D]
  const float* W_qkv = (const float*)d_in[1];  // [3D,D]
  const float* W_out = (const float*)d_in[2];  // [D,D]
  float* out = (float*)d_out;                  // [B,S,D] fp32

  // workspace layout (bytes), everything 256B aligned; total 72 MiB
  char* ws = (char*)d_ws;
  bf16* xb    = (bf16*)(ws);                               //  8 MiB  [B*S, D]
  bf16* wqkvb = (bf16*)(ws + 8u  * 1024 * 1024);           //  6 MiB  [3D, D]
  bf16* woutb = (bf16*)(ws + 14u * 1024 * 1024);           //  2 MiB  [D, D]
  bf16* qkvb  = (bf16*)(ws + 16u * 1024 * 1024);           // 24 MiB  [B*S, 3D]
  bf16* Qhb   = (bf16*)(ws + 40u * 1024 * 1024);           //  8 MiB  [B,H,S,DK]
  bf16* Khb   = (bf16*)(ws + 48u * 1024 * 1024);           //  8 MiB  [B,H,S,DK]
  bf16* Vtb   = (bf16*)(ws + 56u * 1024 * 1024);           //  8 MiB  [B,H,DK,S]
  bf16* attnb = (bf16*)(ws + 64u * 1024 * 1024);           //  8 MiB  [B*S, D]

  const int M = BATCH * S_LEN;   // 4096

  // 1) convert operands to bf16
  cvt_f32_to_bf16_x4<<<(M * DMODEL) / 1024, 256, 0, stream>>>(x, xb);
  cvt_f32_to_bf16_x4<<<(D3 * DMODEL) / 1024, 256, 0, stream>>>(W_qkv, wqkvb);
  cvt_f32_to_bf16_x4<<<(DMODEL * DMODEL) / 1024, 256, 0, stream>>>(W_out, woutb);

  // 2) qkv = x @ W_qkv^T  (bf16 in/out, fp32 accumulate)
  gemm_bf16_rowB<false><<<dim3(M / 128, D3 / 128), 256, 0, stream>>>(
      xb, wqkvb, qkvb, M, D3, DMODEL);

  // 3) RoPE + split into head-major Q/K and transposed V
  rope_split_kernel<<<(BATCH * S_LEN * NHEADS * 32) / 256, 256, 0, stream>>>(
      qkvb, Qhb, Khb, Vtb);

  // 4) causal flash attention (wave per 32-query tile, strided for balance)
  flash_attn_kernel<<<BATCH * NHEADS * 8, 256, 0, stream>>>(
      Qhb, Khb, Vtb, attnb);

  // 5) out = attn @ W_out^T  (fp32 stores to d_out)
  gemm_bf16_rowB<true><<<dim3(M / 128, DMODEL / 128), 256, 0, stream>>>(
      attnb, woutb, out, M, DMODEL, DMODEL);
}
